// MulAttCNN_42228118454861
// MI455X (gfx1250) — matrix operand order repaired and checked by reading, not compiled, not run on hardware
//
#include <hip/hip_runtime.h>
#include <hip/hip_bf16.h>

// ---------------------------------------------------------------------------
// MulAttCNN on MI455X (gfx1250, wave32).
// Dominant cost: conv1d as GEMM  M=32768, N=512, K=3600 -> 120.8 GFLOP via
// v_wmma_f32_16x16x32_bf16. K padded per shift to 1216 = 38*32 so the flat
// 114-step K loop has perfectly linear A/B addresses -> register double
// buffering (software pipeline) instead of load->wait0->wmma serialization.
// All other stages are <4 GFLOP / bandwidth-trivial (~300 MB @ 23.3 TB/s).
// ---------------------------------------------------------------------------

#define BB   128
#define LL   256
#define EE   300
#define PFF  50
#define DD   400          // E + 2*PF
#define D3C  1200
#define D3P  1216         // padded to 38*32
#define CC   512
#define RR   53
#define RP   64           // R padded to 4 WMMA N-tiles
#define KSTEPS 38         // K steps per conv shift (1216/32)
#define KT_TOTAL 114      // 3 shifts * 38
#define NT_CONV 32        // 512/16 N tiles
#define LROWS (LL + 2)    // zero-padded sequence rows
#define BSTRIDE_CONV 16384  // elements per kt in packed conv B (32 nt * 32 ln * 16)
#define BSTRIDE_P    2048   // elements per kt in packed rela_w B (4 * 32 * 16)

typedef __attribute__((ext_vector_type(16))) __bf16 v16bf;
typedef __attribute__((ext_vector_type(8)))  float  v8f;

union FragU { uint4 u[2]; v16bf v; };

static __device__ __forceinline__ float bf2f(__hip_bfloat16 h) { return __bfloat162float(h); }
static __device__ __forceinline__ __hip_bfloat16 f2bf(float f) { return __float2bfloat16(f); }

// 16 contiguous bf16 (two b128 loads) -> fragment register pair
static __device__ __forceinline__ v16bf ld16bf(const __hip_bfloat16* p) {
    FragU f;
    f.u[0] = *(const uint4*)(p);
    f.u[1] = *(const uint4*)(p + 8);
    return f.v;
}
// A fragment halves are 16 elements apart (K0..7|K16..23 per lane group)
static __device__ __forceinline__ v16bf ld_a(const __hip_bfloat16* p) {
    FragU f;
    f.u[0] = *(const uint4*)(p);
    f.u[1] = *(const uint4*)(p + 16);
    return f.v;
}

#define WMMA_BF16(ACC, A, Bf) \
    ACC = __builtin_amdgcn_wmma_f32_16x16x32_bf16(false, (A), false, (Bf), (short)0, (ACC), false, false)

// ---------------------------------------------------------------------------
// K1: gather embeddings + position features -> wfP (B, L+2, 400) bf16, rows
// 0 and L+1 zeroed.
// ---------------------------------------------------------------------------
__global__ void k_gather(const int* __restrict__ context,
                         const int* __restrict__ sdis,
                         const int* __restrict__ odis,
                         const float* __restrict__ embed,
                         const float* __restrict__ pos,
                         __hip_bfloat16* __restrict__ wfP) {
    int row = blockIdx.x % LROWS;
    int b   = blockIdx.x / LROWS;
    __hip_bfloat16* dst = wfP + ((size_t)b * LROWS + row) * DD;
    if (row == 0 || row == LROWS - 1) {
        for (int i = threadIdx.x; i < DD; i += blockDim.x) dst[i] = f2bf(0.0f);
        return;
    }
    int l   = row - 1;
    int tok = context[b * LL + l];
    int sd  = sdis[b * LL + l];
    int od  = odis[b * LL + l];
    for (int i = threadIdx.x; i < DD; i += blockDim.x) {
        float v;
        if (i < EE)            v = embed[(size_t)tok * EE + i];
        else if (i < EE + PFF) v = pos[(size_t)sd * PFF + (i - EE)];
        else                   v = pos[(size_t)od * PFF + (i - EE - PFF)];
        dst[i] = f2bf(v);
    }
}

// ---------------------------------------------------------------------------
// K2: span means -> subjobj (B, 2, 1200) fp32.
// ---------------------------------------------------------------------------
__global__ void k_spans(const int* __restrict__ sidx,
                        const int* __restrict__ oidx,
                        const __hip_bfloat16* __restrict__ wfP,
                        float* __restrict__ subjobj) {
    int b = blockIdx.x;
    const __hip_bfloat16* W = wfP + (size_t)b * LROWS * DD;
    for (int which = 0; which < 2; ++which) {
        const int* idx = which ? oidx : sidx;
        int s0 = idx[b * 2 + 0], s1 = idx[b * 2 + 1];
        float inv = 1.0f / (float)(s1 - s0 + 1);
        float* out = subjobj + (size_t)b * 2 * D3C + which * D3C;
        for (int jd = threadIdx.x; jd < D3C; jd += blockDim.x) {
            int j = jd / DD, d = jd - j * DD;
            float acc = 0.0f;
            for (int m = s0; m <= s1; ++m)
                acc += bf2f(W[(size_t)(m + j) * DD + d]);
            out[jd] = acc * inv;
        }
    }
}

// ---------------------------------------------------------------------------
// K3a: logits[b,l,{s,o}] = wf3[b,l] . subj/obj
// ---------------------------------------------------------------------------
__global__ void k_logits(const __hip_bfloat16* __restrict__ wfP,
                         const float* __restrict__ subjobj,
                         float* __restrict__ logits) {
    int l = blockIdx.x & (LL - 1);
    int b = blockIdx.x >> 8;
    const __hip_bfloat16* W = wfP + (size_t)b * LROWS * DD;
    const float* subj = subjobj + (size_t)b * 2 * D3C;
    const float* obj  = subj + D3C;
    float ps = 0.0f, po = 0.0f;
    for (int jd = threadIdx.x; jd < D3C; jd += blockDim.x) {
        int j = jd / DD, d = jd - j * DD;
        float w = bf2f(W[(size_t)(l + j) * DD + d]);
        ps += w * subj[jd];
        po += w * obj[jd];
    }
    __shared__ float r0[128], r1[128];
    r0[threadIdx.x] = ps; r1[threadIdx.x] = po;
    __syncthreads();
    for (int s = 64; s > 0; s >>= 1) {
        if (threadIdx.x < s) { r0[threadIdx.x] += r0[threadIdx.x + s]; r1[threadIdx.x] += r1[threadIdx.x + s]; }
        __syncthreads();
    }
    if (threadIdx.x == 0) {
        logits[((size_t)b * LL + l) * 2 + 0] = r0[0];
        logits[((size_t)b * LL + l) * 2 + 1] = r1[0];
    }
}

// ---------------------------------------------------------------------------
// K3b: per-b softmax over l -> scale = 0.5*(rs+ro)
// ---------------------------------------------------------------------------
__global__ void k_softscale(const float* __restrict__ logits,
                            float* __restrict__ scale) {
    int b = blockIdx.x, l = threadIdx.x;
    float vs = logits[((size_t)b * LL + l) * 2 + 0];
    float vo = logits[((size_t)b * LL + l) * 2 + 1];
    __shared__ float sm[LL];
    sm[l] = vs; __syncthreads();
    for (int s = 128; s > 0; s >>= 1) { if (l < s) sm[l] = fmaxf(sm[l], sm[l + s]); __syncthreads(); }
    float ms = sm[0]; __syncthreads();
    sm[l] = expf(vs - ms); __syncthreads();
    for (int s = 128; s > 0; s >>= 1) { if (l < s) sm[l] += sm[l + s]; __syncthreads(); }
    float ss = sm[0]; __syncthreads();
    sm[l] = vo; __syncthreads();
    for (int s = 128; s > 0; s >>= 1) { if (l < s) sm[l] = fmaxf(sm[l], sm[l + s]); __syncthreads(); }
    float mo = sm[0]; __syncthreads();
    sm[l] = expf(vo - mo); __syncthreads();
    for (int s = 128; s > 0; s >>= 1) { if (l < s) sm[l] += sm[l + s]; __syncthreads(); }
    float so = sm[0];
    scale[(size_t)b * LL + l] = 0.5f * (expf(vs - ms) / ss + expf(vo - mo) / so);
}

// ---------------------------------------------------------------------------
// K4: scaled window tensor wf3s (B, L+2, 1216) bf16
// ---------------------------------------------------------------------------
__global__ void k_wf3s(const __hip_bfloat16* __restrict__ wfP,
                       const float* __restrict__ scale,
                       __hip_bfloat16* __restrict__ wf3s) {
    int mp = blockIdx.x % LROWS;
    int b  = blockIdx.x / LROWS;
    __hip_bfloat16* dst = wf3s + ((size_t)b * LROWS + mp) * D3P;
    if (mp == 0 || mp == LROWS - 1) {
        for (int i = threadIdx.x; i < D3P; i += blockDim.x) dst[i] = f2bf(0.0f);
        return;
    }
    float s = scale[(size_t)b * LL + (mp - 1)];
    const __hip_bfloat16* W = wfP + (size_t)b * LROWS * DD;
    for (int i = threadIdx.x; i < D3P; i += blockDim.x) {
        float v = 0.0f;
        if (i < D3C) {
            int j = i / DD, d = i - j * DD;
            v = s * bf2f(W[(size_t)(mp + j - 1) * DD + d]);
        }
        dst[i] = f2bf(v);
    }
}

// ---------------------------------------------------------------------------
// K5a: pack conv weights into B-fragment layout.
// ---------------------------------------------------------------------------
__global__ void k_wtpack(const float* __restrict__ conv_w,
                         __hip_bfloat16* __restrict__ wtp) {
    long id = (long)blockIdx.x * blockDim.x + threadIdx.x;
    if (id >= (long)KT_TOTAL * NT_CONV * 32 * 16) return;
    int j  = (int)(id & 15);
    int ln = (int)((id >> 4) & 31);
    int nt = (int)((id >> 9) & 31);
    int kt = (int)(id >> 14);
    int shift = kt / KSTEPS, step = kt - shift * KSTEPS;
    int kk = step * 32 + ((ln >> 4) << 3) + (j & 7) + ((j >> 3) << 4);
    int c  = nt * 16 + (ln & 15);
    float v = (kk < D3C) ? conv_w[(size_t)c * (D3C * 3) + (size_t)kk * 3 + shift] : 0.0f;
    wtp[(((size_t)kt * NT_CONV + nt) * 32 + ln) * 16 + j] = f2bf(v);
}

// ---------------------------------------------------------------------------
// K5b: pack rela_linear_w (53,512) as B: B[k=c, n=r], r padded to 64.
// ---------------------------------------------------------------------------
__global__ void k_rwpack(const float* __restrict__ rw,
                         __hip_bfloat16* __restrict__ rwp) {
    int id = blockIdx.x * blockDim.x + threadIdx.x;  // 16*4*32*16 = 32768
    int j  = id & 15;
    int ln = (id >> 4) & 31;
    int nt = (id >> 9) & 3;
    int kt = id >> 11;
    int K = kt * 32 + ((ln >> 4) << 3) + (j & 7) + ((j >> 3) << 4);
    int N = nt * 16 + (ln & 15);
    float v = (N < RR) ? rw[(size_t)N * CC + K] : 0.0f;
    rwp[(((size_t)kt * 4 + nt) * 32 + ln) * 16 + j] = f2bf(v);
}

// ---------------------------------------------------------------------------
// K6: conv as WMMA GEMM, flat 114-step K loop, register double buffered.
// Wave tile 16(M) x 64(N); workgroup = 8 waves; grid = B*16.
// A address is linear in flat kt: base + kt*32 elements (D3P == 38*32 makes
// the shift transition seamless). Packed B advances 16384 elements per kt.
// Tail (kt==114) loads read unused in-workspace garbage (regions padded).
// ---------------------------------------------------------------------------
__global__ void __launch_bounds__(256) k_convgemm(
        const __hip_bfloat16* __restrict__ wf3s,
        const __hip_bfloat16* __restrict__ wtp,
        const float* __restrict__ conv_b,
        __hip_bfloat16* __restrict__ sent) {
    int wg    = blockIdx.x;
    int b     = wg >> 4;
    int t     = wg & 15;
    int mtile = t & 7;
    int ntile = t >> 3;
    int wave  = threadIdx.x >> 5;
    int lane  = threadIdx.x & 31;
    int mw    = wave & 1;
    int nw    = wave >> 1;
    int row0  = mtile * 32 + mw * 16;
    int n0    = ntile * 256 + nw * 64;

    int m  = lane & 15;
    int ko = (lane & 16) ? 8 : 0;
    const __hip_bfloat16* pa = wf3s + (size_t)b * LROWS * D3P + (size_t)(row0 + m) * D3P + ko;
    const __hip_bfloat16* pb = wtp + (((size_t)(n0 >> 4) * 32) + lane) * 16;

    v8f acc0 = {}, acc1 = {}, acc2 = {}, acc3 = {};
    v16bf a0, a1, b00, b01, b02, b03, b10, b11, b12, b13;

    a0  = ld_a(pa);
    b00 = ld16bf(pb);
    b01 = ld16bf(pb + 512);
    b02 = ld16bf(pb + 1024);
    b03 = ld16bf(pb + 1536);

    for (int kt = 0; kt < KT_TOTAL; kt += 2) {
        const __hip_bfloat16* pa1 = pa + (size_t)(kt + 1) * 32;
        const __hip_bfloat16* pb1 = pb + (size_t)(kt + 1) * BSTRIDE_CONV;
        a1  = ld_a(pa1);
        b10 = ld16bf(pb1);
        b11 = ld16bf(pb1 + 512);
        b12 = ld16bf(pb1 + 1024);
        b13 = ld16bf(pb1 + 1536);
        __builtin_prefetch(pa + (size_t)(kt + 4) * 32, 0, 1);
        WMMA_BF16(acc0, a0, b00);
        WMMA_BF16(acc1, a0, b01);
        WMMA_BF16(acc2, a0, b02);
        WMMA_BF16(acc3, a0, b03);
        const __hip_bfloat16* pa2 = pa + (size_t)(kt + 2) * 32;
        const __hip_bfloat16* pb2 = pb + (size_t)(kt + 2) * BSTRIDE_CONV;
        a0  = ld_a(pa2);                 // kt==112 -> loads kt=114: unused garbage
        b00 = ld16bf(pb2);
        b01 = ld16bf(pb2 + 512);
        b02 = ld16bf(pb2 + 1024);
        b03 = ld16bf(pb2 + 1536);
        WMMA_BF16(acc0, a1, b10);
        WMMA_BF16(acc1, a1, b11);
        WMMA_BF16(acc2, a1, b12);
        WMMA_BF16(acc3, a1, b13);
    }

    int mrow = (lane & 16) ? 8 : 0;
    int col  = lane & 15;
    v8f accs[4] = {acc0, acc1, acc2, acc3};
#pragma unroll
    for (int jj = 0; jj < 4; ++jj) {
        int c = n0 + jj * 16 + col;
        float bias = conv_b[c];
#pragma unroll
        for (int i = 0; i < 8; ++i) {
            int l = row0 + mrow + i;
            sent[((size_t)b * LL + l) * CC + c] = f2bf(tanhf(accs[jj][i] + bias));
        }
    }
}

// ---------------------------------------------------------------------------
// K7: p = sentbf (32768x512) x rela_w^T (512x64pad), same pipelined scheme.
// ---------------------------------------------------------------------------
__global__ void __launch_bounds__(256) k_pgemm(
        const __hip_bfloat16* __restrict__ sent,
        const __hip_bfloat16* __restrict__ rwp,
        __hip_bfloat16* __restrict__ p) {
    int wave = threadIdx.x >> 5;
    int lane = threadIdx.x & 31;
    int row0 = (blockIdx.x * 8 + wave) * 16;

    int m  = lane & 15;
    int ko = (lane & 16) ? 8 : 0;
    const __hip_bfloat16* pa = sent + (size_t)(row0 + m) * CC + ko;
    const __hip_bfloat16* pb = rwp + (size_t)lane * 16;

    v8f acc0 = {}, acc1 = {}, acc2 = {}, acc3 = {};
    v16bf a0, a1, b00, b01, b02, b03, b10, b11, b12, b13;

    a0  = ld_a(pa);
    b00 = ld16bf(pb);
    b01 = ld16bf(pb + 512);
    b02 = ld16bf(pb + 1024);
    b03 = ld16bf(pb + 1536);

    for (int kt = 0; kt < 16; kt += 2) {
        const __hip_bfloat16* pa1 = pa + (size_t)(kt + 1) * 32;
        const __hip_bfloat16* pb1 = pb + (size_t)(kt + 1) * BSTRIDE_P;
        a1  = ld_a(pa1);
        b10 = ld16bf(pb1);
        b11 = ld16bf(pb1 + 512);
        b12 = ld16bf(pb1 + 1024);
        b13 = ld16bf(pb1 + 1536);
        WMMA_BF16(acc0, a0, b00);
        WMMA_BF16(acc1, a0, b01);
        WMMA_BF16(acc2, a0, b02);
        WMMA_BF16(acc3, a0, b03);
        const __hip_bfloat16* pa2 = pa + (size_t)(kt + 2) * 32;
        const __hip_bfloat16* pb2 = pb + (size_t)(kt + 2) * BSTRIDE_P;
        a0  = ld_a(pa2);                 // kt==14 -> loads kt=16: unused garbage
        b00 = ld16bf(pb2);
        b01 = ld16bf(pb2 + 512);
        b02 = ld16bf(pb2 + 1024);
        b03 = ld16bf(pb2 + 1536);
        WMMA_BF16(acc0, a1, b10);
        WMMA_BF16(acc1, a1, b11);
        WMMA_BF16(acc2, a1, b12);
        WMMA_BF16(acc3, a1, b13);
    }

    int mrow = (lane & 16) ? 8 : 0;
    int col  = lane & 15;
    v8f accs[4] = {acc0, acc1, acc2, acc3};
#pragma unroll
    for (int jj = 0; jj < 4; ++jj)
#pragma unroll
        for (int i = 0; i < 8; ++i)
            p[(size_t)(row0 + mrow + i) * RP + jj * 16 + col] = f2bf(accs[jj][i]);
}

// ---------------------------------------------------------------------------
// K8: fused attention tail per batch.
// ---------------------------------------------------------------------------
__global__ void k_attn(const __hip_bfloat16* __restrict__ p,
                       const __hip_bfloat16* __restrict__ sent,
                       const float* __restrict__ re,
                       float* __restrict__ wo) {
    int b = blockIdx.x;
    __shared__ __hip_bfloat16 pl[LL * RP];       // 32 KB
    {
        const uint4* src = (const uint4*)(p + (size_t)b * LL * RP);
        uint4* dst = (uint4*)pl;
        for (int i = threadIdx.x; i < LL * RP * 2 / 16; i += blockDim.x) dst[i] = src[i];
    }
    __syncthreads();

    for (int c = threadIdx.x; c < CC; c += blockDim.x) {
        float rc[RR];
#pragma unroll
        for (int r = 0; r < RR; ++r) rc[r] = re[(size_t)r * CC + c];
        float qmax = -3.4e38f;
        for (int l = 0; l < LL; ++l) {
            float q = 0.0f;
#pragma unroll
            for (int r = 0; r < RR; ++r) q += bf2f(pl[l * RP + r]) * rc[r];
            qmax = fmaxf(qmax, q);
        }
        float Z = 0.0f, tmax = -3.4e38f;
        for (int l = 0; l < LL; ++l) {
            float q = 0.0f;
#pragma unroll
            for (int r = 0; r < RR; ++r) q += bf2f(pl[l * RP + r]) * rc[r];
            float e = expf(q - qmax);
            Z += e;
            float s = bf2f(sent[((size_t)b * LL + l) * CC + c]);
            tmax = fmaxf(tmax, s * e);
        }
        wo[(size_t)b * CC + c] = tmax / Z;
    }
}

// ---------------------------------------------------------------------------
// K9: normalize wo, distances to rela_emb rows -> (B, 53) fp32.
// ---------------------------------------------------------------------------
__global__ void k_dist(const float* __restrict__ wo,
                       const float* __restrict__ re,
                       float* __restrict__ out) {
    int b = blockIdx.x;
    __shared__ float sm[256];
    float acc = 0.0f;
    for (int c = threadIdx.x; c < CC; c += blockDim.x) {
        float v = wo[(size_t)b * CC + c];
        acc += v * v;
    }
    sm[threadIdx.x] = acc; __syncthreads();
    for (int s = 128; s > 0; s >>= 1) {
        if (threadIdx.x < s) sm[threadIdx.x] += sm[threadIdx.x + s];
        __syncthreads();
    }
    float nrm = fmaxf(sqrtf(sm[0]), 1e-12f);
    if (threadIdx.x < RR) {
        int r = threadIdx.x;
        float d2 = 0.0f;
        for (int c = 0; c < CC; ++c) {
            float d = wo[(size_t)b * CC + c] / nrm - re[(size_t)r * CC + c];
            d2 += d * d;
        }
        out[(size_t)b * RR + r] = sqrtf(d2);
    }
}

// ---------------------------------------------------------------------------
// Host launch
// ---------------------------------------------------------------------------
extern "C" void kernel_launch(void* const* d_in, const int* in_sizes, int n_in,
                              void* d_out, int out_size, void* d_ws, size_t ws_size,
                              hipStream_t stream) {
    (void)in_sizes; (void)n_in; (void)out_size; (void)ws_size;
    const int*   context = (const int*)d_in[0];
    const int*   sidx    = (const int*)d_in[1];
    const int*   oidx    = (const int*)d_in[2];
    const int*   sdis    = (const int*)d_in[3];
    const int*   odis    = (const int*)d_in[4];
    const float* embed   = (const float*)d_in[5];
    const float* pos     = (const float*)d_in[6];
    const float* conv_w  = (const float*)d_in[7];
    const float* conv_b  = (const float*)d_in[8];
    const float* rela_w  = (const float*)d_in[9];
    const float* rela_e  = (const float*)d_in[10];
    float* out = (float*)d_out;

    // Workspace layout (bytes, 256-aligned; each region padded so the GEMMs'
    // one-past-the-end pipeline tail loads stay inside the workspace).
    char* ws = (char*)d_ws;
    constexpr size_t OFF_WFP  = 0;            // (B,258,400)  bf16  26,419,200
    constexpr size_t OFF_SO   = 26419200;     // (B,2,1200)   f32    1,228,800
    constexpr size_t OFF_LG   = 27648000;     // (B,L,2)      f32      262,144
    constexpr size_t OFF_SC   = 27910144;     // (B,L)        f32      131,072
    constexpr size_t OFF_W3   = 28041216;     // (B,258,1216) bf16  80,314,368 (+pad)
    constexpr size_t OFF_WTP  = 108359680;    // packed conv W       3,735,552 (+pad)
    constexpr size_t OFF_SENT = 112160768;    // (B*L,512)    bf16  33,554,432 (+pad)
    constexpr size_t OFF_RWP  = 145780736;    // packed rela_w          65,536 (+pad)
    constexpr size_t OFF_P    = 145911808;    // (B*L,64)     bf16   4,194,304
    constexpr size_t OFF_WO   = 150106112;    // (B,512)      f32      262,144

    __hip_bfloat16* wfP  = (__hip_bfloat16*)(ws + OFF_WFP);
    float*          sobj = (float*)(ws + OFF_SO);
    float*          lg   = (float*)(ws + OFF_LG);
    float*          sc   = (float*)(ws + OFF_SC);
    __hip_bfloat16* w3   = (__hip_bfloat16*)(ws + OFF_W3);
    __hip_bfloat16* wtp  = (__hip_bfloat16*)(ws + OFF_WTP);
    __hip_bfloat16* sent = (__hip_bfloat16*)(ws + OFF_SENT);
    __hip_bfloat16* rwp  = (__hip_bfloat16*)(ws + OFF_RWP);
    __hip_bfloat16* pbuf = (__hip_bfloat16*)(ws + OFF_P);
    float*          wo   = (float*)(ws + OFF_WO);

    // weight packing (cheap; rerun every call for determinism)
    k_wtpack<<<dim3((KT_TOTAL * NT_CONV * 32 * 16 + 255) / 256), dim3(256), 0, stream>>>(conv_w, wtp);
    k_rwpack<<<dim3(128), dim3(256), 0, stream>>>(rela_w, rwp);

    // feature build
    k_gather<<<dim3(BB * LROWS), dim3(128), 0, stream>>>(context, sdis, odis, embed, pos, wfP);
    k_spans<<<dim3(BB), dim3(256), 0, stream>>>(sidx, oidx, wfP, sobj);
    k_logits<<<dim3(BB * LL), dim3(128), 0, stream>>>(wfP, sobj, lg);
    k_softscale<<<dim3(BB), dim3(LL), 0, stream>>>(lg, sc);
    k_wf3s<<<dim3(BB * LROWS), dim3(256), 0, stream>>>(wfP, sc, w3);

    // heavy WMMA GEMMs (software pipelined)
    k_convgemm<<<dim3(BB * 16), dim3(256), 0, stream>>>(w3, wtp, conv_b, sent);
    k_pgemm<<<dim3(256), dim3(256), 0, stream>>>(sent, rwp, pbuf);

    // fused tail
    k_attn<<<dim3(BB), dim3(256), 0, stream>>>(pbuf, sent, rela_e, wo);
    k_dist<<<dim3(BB), dim3(256), 0, stream>>>(wo, rela_e, out);
}
